// DualSoftMaxMatcherV2_50826642981154
// MI455X (gfx1250) — compile-verified, matrix operand order checked by reading, not run on hardware
//
#include <hip/hip_runtime.h>
#include <hip/hip_bf16.h>
#include <math.h>

#define BATCH     2
#define NROWS     4096
#define MCOLS     4096
#define DDIM      256
#define INV_TEMP  20.0f
#define THRESHOLD 0.01f

typedef __attribute__((ext_vector_type(16))) __bf16 v16bf;
typedef __attribute__((ext_vector_type(8)))  float  v8f;

union FragU { v16bf v; uint4 q[2]; };

static __device__ __forceinline__ unsigned short f32_to_bf16_rne(float f) {
    unsigned int u = __float_as_uint(f);
    unsigned int r = (u + 0x7FFFu + ((u >> 16) & 1u)) >> 16;
    return (unsigned short)r;
}

// ---------------------------------------------------------------------------
// 1) L2-normalize each 256-dim descriptor row, convert to bf16.
//    One wave (32 lanes) per row; 8 floats per lane.
// ---------------------------------------------------------------------------
__global__ __launch_bounds__(256)
void k_normalize(const float* __restrict__ src, unsigned short* __restrict__ dst, int rows) {
    int wave = (blockIdx.x * blockDim.x + threadIdx.x) >> 5;
    int lane = threadIdx.x & 31;
    if (wave >= rows) return;
    const float* p = src + (size_t)wave * DDIM + lane * 8;
    float4 a = ((const float4*)p)[0];
    float4 b = ((const float4*)p)[1];
    float ss = a.x*a.x + a.y*a.y + a.z*a.z + a.w*a.w
             + b.x*b.x + b.y*b.y + b.z*b.z + b.w*b.w;
    #pragma unroll
    for (int off = 16; off > 0; off >>= 1) ss += __shfl_xor(ss, off, 32);
    float inv = rsqrtf(ss + 1e-8f);
    union { unsigned short h[8]; uint4 q; } o;
    o.h[0] = f32_to_bf16_rne(a.x * inv);
    o.h[1] = f32_to_bf16_rne(a.y * inv);
    o.h[2] = f32_to_bf16_rne(a.z * inv);
    o.h[3] = f32_to_bf16_rne(a.w * inv);
    o.h[4] = f32_to_bf16_rne(b.x * inv);
    o.h[5] = f32_to_bf16_rne(b.y * inv);
    o.h[6] = f32_to_bf16_rne(b.z * inv);
    o.h[7] = f32_to_bf16_rne(b.w * inv);
    *(uint4*)(dst + (size_t)wave * DDIM + lane * 8) = o.q;
}

// ---------------------------------------------------------------------------
// 2) NT GEMM via v_wmma_f32_16x16x32_bf16, register-blocked.
//    Block = 256 threads = 8 waves arranged 2x4; block tile = 64x128.
//    Each wave computes a 32x32 tile: 2 A frags x 2 B frags -> 4 WMMAs per
//    K=32 step (2 b128 loads per WMMA instead of 4).
//    Fragment layout (16-bit A 16x32): lane l (row = l&15, h = l>>4) holds
//    k in [32t + 8h, +8) and [32t + 16 + 8h, +8). B frag is identical on
//    dB rows (NT GEMM).
// ---------------------------------------------------------------------------
__global__ __launch_bounds__(256)
void k_gemm_wmma(const unsigned short* __restrict__ dA,
                 const unsigned short* __restrict__ dB,
                 float* __restrict__ sim) {
    int lane = threadIdx.x & 31;
    int wave = threadIdx.x >> 5;
    int wi = wave >> 2;        // 0..1
    int wj = wave & 3;         // 0..3
    int b  = blockIdx.z;
    int i0 = blockIdx.y * 64  + wi * 32;
    int j0 = blockIdx.x * 128 + wj * 32;
    int r = lane & 15;
    int h = lane >> 4;

    const unsigned short* arow0 = dA + ((size_t)b * NROWS + i0 + r) * DDIM;
    const unsigned short* arow1 = arow0 + (size_t)16 * DDIM;
    const unsigned short* brow0 = dB + ((size_t)b * MCOLS + j0 + r) * DDIM;
    const unsigned short* brow1 = brow0 + (size_t)16 * DDIM;

    v8f c00 = {}, c01 = {}, c10 = {}, c11 = {};
    #pragma unroll
    for (int kt = 0; kt < 8; ++kt) {
        int k0 = kt * 32 + h * 8;
        FragU fa0, fa1, fb0, fb1;
        fa0.q[0] = *(const uint4*)(arow0 + k0);
        fa0.q[1] = *(const uint4*)(arow0 + k0 + 16);
        fa1.q[0] = *(const uint4*)(arow1 + k0);
        fa1.q[1] = *(const uint4*)(arow1 + k0 + 16);
        fb0.q[0] = *(const uint4*)(brow0 + k0);
        fb0.q[1] = *(const uint4*)(brow0 + k0 + 16);
        fb1.q[0] = *(const uint4*)(brow1 + k0);
        fb1.q[1] = *(const uint4*)(brow1 + k0 + 16);
        c00 = __builtin_amdgcn_wmma_f32_16x16x32_bf16(
                  false, fa0.v, false, fb0.v, (short)0, c00, false, false);
        c01 = __builtin_amdgcn_wmma_f32_16x16x32_bf16(
                  false, fa0.v, false, fb1.v, (short)0, c01, false, false);
        c10 = __builtin_amdgcn_wmma_f32_16x16x32_bf16(
                  false, fa1.v, false, fb0.v, (short)0, c10, false, false);
        c11 = __builtin_amdgcn_wmma_f32_16x16x32_bf16(
                  false, fa1.v, false, fb1.v, (short)0, c11, false, false);
    }

    // C/D layout: VGPR v, lane l -> row = v + 8*(l>>4), col = l&15
    size_t rowBase = (size_t)b * NROWS + i0 + h * 8;
    int    col     = lane & 15;
    {
        float* o = sim + rowBase * (size_t)MCOLS + j0 + col;
        #pragma unroll
        for (int v = 0; v < 8; ++v) o[(size_t)v * MCOLS] = c00[v] * INV_TEMP;
    }
    {
        float* o = sim + rowBase * (size_t)MCOLS + j0 + 16 + col;
        #pragma unroll
        for (int v = 0; v < 8; ++v) o[(size_t)v * MCOLS] = c01[v] * INV_TEMP;
    }
    {
        float* o = sim + (rowBase + 16) * (size_t)MCOLS + j0 + col;
        #pragma unroll
        for (int v = 0; v < 8; ++v) o[(size_t)v * MCOLS] = c10[v] * INV_TEMP;
    }
    {
        float* o = sim + (rowBase + 16) * (size_t)MCOLS + j0 + 16 + col;
        #pragma unroll
        for (int v = 0; v < 8; ++v) o[(size_t)v * MCOLS] = c11[v] * INV_TEMP;
    }
}

// ---------------------------------------------------------------------------
// 3) Row softmax stats: max + sum(exp). One block (256 thr) per row,
//    16 f32 per thread held in registers (single pass over memory).
// ---------------------------------------------------------------------------
__global__ __launch_bounds__(256)
void k_row_stats(const float* __restrict__ sim,
                 float* __restrict__ rmax, float* __restrict__ rsum) {
    __shared__ float red[256];
    int row = blockIdx.x;
    int tid = threadIdx.x;
    const float4* p = (const float4*)(sim + (size_t)row * MCOLS);
    float4 x[4];
    #pragma unroll
    for (int t = 0; t < 4; ++t) x[t] = p[tid + t * 256];

    float m = -INFINITY;
    #pragma unroll
    for (int t = 0; t < 4; ++t) {
        m = fmaxf(m, fmaxf(fmaxf(x[t].x, x[t].y), fmaxf(x[t].z, x[t].w)));
    }
    red[tid] = m; __syncthreads();
    for (int s = 128; s > 0; s >>= 1) {
        if (tid < s) red[tid] = fmaxf(red[tid], red[tid + s]);
        __syncthreads();
    }
    m = red[0]; __syncthreads();

    float sum = 0.f;
    #pragma unroll
    for (int t = 0; t < 4; ++t) {
        sum += __expf(x[t].x - m) + __expf(x[t].y - m)
             + __expf(x[t].z - m) + __expf(x[t].w - m);
    }
    red[tid] = sum; __syncthreads();
    for (int s = 128; s > 0; s >>= 1) {
        if (tid < s) red[tid] += red[tid + s];
        __syncthreads();
    }
    if (tid == 0) { rmax[row] = m; rsum[row] = red[0]; }
}

// ---------------------------------------------------------------------------
// 4) Column softmax stats via streaming online-softmax (single pass,
//    coalesced: thread owns a column).
// ---------------------------------------------------------------------------
__global__ __launch_bounds__(256)
void k_col_stats(const float* __restrict__ sim,
                 float* __restrict__ cmax, float* __restrict__ csum) {
    int b = blockIdx.y;
    int j = blockIdx.x * blockDim.x + threadIdx.x;
    const float* p = sim + (size_t)b * NROWS * MCOLS + j;
    float m = -INFINITY, s = 0.f;
    for (int i = 0; i < NROWS; ++i) {
        float x  = p[(size_t)i * MCOLS];
        float mn = fmaxf(m, x);
        s = s * __expf(m - mn) + __expf(x - mn);
        m = mn;
    }
    cmax[b * MCOLS + j] = m;
    csum[b * MCOLS + j] = s;
}

// ---------------------------------------------------------------------------
// 5) P = softmax_row * softmax_col, computed column-wise in place;
//    track per-column max(P).
// ---------------------------------------------------------------------------
__global__ __launch_bounds__(256)
void k_p_col(float* __restrict__ P,
             const float* __restrict__ rmax, const float* __restrict__ rsum,
             const float* __restrict__ cmax, const float* __restrict__ csum,
             float* __restrict__ colPmax) {
    int b = blockIdx.y;
    int j = blockIdx.x * blockDim.x + threadIdx.x;
    float cm = cmax[b * MCOLS + j];
    float ci = 1.f / csum[b * MCOLS + j];
    float* p = P + (size_t)b * NROWS * MCOLS + j;
    const float* rm = rmax + (size_t)b * NROWS;
    const float* rs = rsum + (size_t)b * NROWS;
    float best = 0.f;
    for (int i = 0; i < NROWS; ++i) {
        float sv = p[(size_t)i * MCOLS];
        float pv = __expf(sv - rm[i]) * (1.f / rs[i]) * __expf(sv - cm) * ci;
        p[(size_t)i * MCOLS] = pv;
        best = fmaxf(best, pv);
    }
    colPmax[b * MCOLS + j] = best;
}

// ---------------------------------------------------------------------------
// 6) Per-row max(P) + argmax. One block per row.
// ---------------------------------------------------------------------------
__global__ __launch_bounds__(256)
void k_rowp(const float* __restrict__ P,
            float* __restrict__ rowPmax, int* __restrict__ bestj) {
    __shared__ float sv[256];
    __shared__ int   si[256];
    int row = blockIdx.x;
    int tid = threadIdx.x;
    const float4* p = (const float4*)(P + (size_t)row * MCOLS);
    float bm = -INFINITY; int bi = 0;
    #pragma unroll
    for (int t = 0; t < 4; ++t) {
        int base = (tid + t * 256) * 4;
        float4 x = p[tid + t * 256];
        if (x.x > bm) { bm = x.x; bi = base + 0; }
        if (x.y > bm) { bm = x.y; bi = base + 1; }
        if (x.z > bm) { bm = x.z; bi = base + 2; }
        if (x.w > bm) { bm = x.w; bi = base + 3; }
    }
    sv[tid] = bm; si[tid] = bi; __syncthreads();
    for (int s = 128; s > 0; s >>= 1) {
        if (tid < s && sv[tid + s] > sv[tid]) {
            sv[tid] = sv[tid + s]; si[tid] = si[tid + s];
        }
        __syncthreads();
    }
    if (tid == 0) { rowPmax[row] = sv[0]; bestj[row] = si[0]; }
}

// ---------------------------------------------------------------------------
// 7) Mutual-max + threshold -> matches_A, matches_B, valid.
// ---------------------------------------------------------------------------
__global__ __launch_bounds__(256)
void k_final(const float* __restrict__ kpA, const float* __restrict__ kpB,
             const float* __restrict__ rowPmax, const int* __restrict__ bestj,
             const float* __restrict__ colPmax,
             float* __restrict__ mA, float* __restrict__ mB,
             float* __restrict__ validOut) {
    int idx = blockIdx.x * blockDim.x + threadIdx.x;
    if (idx >= BATCH * NROWS) return;
    int b = idx >> 12;
    int j = bestj[idx];
    float pm = rowPmax[idx];
    bool valid = (pm > THRESHOLD) && (colPmax[b * MCOLS + j] == pm);
    mA[2 * idx + 0] = kpA[2 * idx + 0];
    mA[2 * idx + 1] = kpA[2 * idx + 1];
    const float* kb = kpB + ((size_t)b * MCOLS + j) * 2;
    mB[2 * idx + 0] = kb[0];
    mB[2 * idx + 1] = kb[1];
    validOut[idx] = valid ? 1.0f : 0.0f;
}

extern "C" void kernel_launch(void* const* d_in, const int* in_sizes, int n_in,
                              void* d_out, int out_size, void* d_ws, size_t ws_size,
                              hipStream_t stream) {
    const float* kpA   = (const float*)d_in[0];   // [B,N,2]
    const float* descA = (const float*)d_in[1];   // [B,N,256]
    const float* kpB   = (const float*)d_in[2];   // [B,M,2]
    const float* descB = (const float*)d_in[3];   // [B,M,256]

    float* out      = (float*)d_out;
    float* mA       = out;                        // B*N*2 = 16384
    float* mB       = out + 16384;                // 16384
    float* validOut = out + 32768;                // 8192
    float* P        = out + 40960;                // B*N*M = 33554432 (sim -> P in place)

    char* ws = (char*)d_ws;
    unsigned short* dAbf = (unsigned short*)ws;                          // 4 MB
    unsigned short* dBbf = (unsigned short*)(ws + (size_t)4 * 1024 * 1024);
    float* rmax    = (float*)(ws + (size_t)8 * 1024 * 1024);
    float* rsum    = rmax + BATCH * NROWS;
    float* cmax    = rsum + BATCH * NROWS;
    float* csum    = cmax + BATCH * MCOLS;
    float* rowPmax = csum + BATCH * MCOLS;
    float* colPmax = rowPmax + BATCH * NROWS;
    int*   bestj   = (int*)(colPmax + BATCH * MCOLS);

    // 1) normalize + bf16 convert (8192 rows each, 8 waves/block)
    k_normalize<<<dim3(1024), dim3(256), 0, stream>>>(descA, dAbf, BATCH * NROWS);
    k_normalize<<<dim3(1024), dim3(256), 0, stream>>>(descB, dBbf, BATCH * MCOLS);

    // 2) WMMA GEMM: block tile 64x128, wave tile 32x32.
    //    grid = (M/128, N/64, B)
    k_gemm_wmma<<<dim3(32, 64, 2), dim3(256), 0, stream>>>(dAbf, dBbf, P);

    // 3) row softmax stats
    k_row_stats<<<dim3(BATCH * NROWS), dim3(256), 0, stream>>>(P, rmax, rsum);

    // 4) column softmax stats (online)
    k_col_stats<<<dim3(MCOLS / 256, BATCH), dim3(256), 0, stream>>>(P, cmax, csum);

    // 5) P in place + per-column max(P)
    k_p_col<<<dim3(MCOLS / 256, BATCH), dim3(256), 0, stream>>>(
        P, rmax, rsum, cmax, csum, colPmax);

    // 6) per-row max(P) + argmax
    k_rowp<<<dim3(BATCH * NROWS), dim3(256), 0, stream>>>(P, rowPmax, bestj);

    // 7) finalize matches
    k_final<<<dim3((BATCH * NROWS + 255) / 256), dim3(256), 0, stream>>>(
        kpA, kpB, rowPmax, bestj, colPmax, mA, mB, validOut);
}
